// ShiftedWindowAttention_88141318849043
// MI455X (gfx1250) — matrix operand order repaired and checked by reading, not compiled, not run on hardware
//
#include <hip/hip_runtime.h>
#include <math.h>

// ---------------------------------------------------------------------------
// Shifted-window attention (Swin) for MI455X / gfx1250, bf16 WMMA pipeline.
//   B=32, H=W=56 (L=3136), C=384, HEADS=12, hd=32, WS=7, M=49, nW=64, SS=3
// ---------------------------------------------------------------------------

typedef __attribute__((ext_vector_type(16))) __bf16       bf16x16;
typedef __attribute__((ext_vector_type(8)))  float        f32x8;
typedef __attribute__((ext_vector_type(4)))  unsigned int u32x4;
typedef __attribute__((ext_vector_type(8)))  int          i32x8;
typedef __attribute__((ext_vector_type(4)))  int          i32x4;
typedef __attribute__((ext_vector_type(2)))  float        f32x2;

union FragU { bf16x16 v; u32x4 u[2]; };
union Bf8U  { u32x4 v; __bf16 h[8]; };

__device__ __forceinline__ unsigned int pack_bf16(float a, float b) {
    union { __bf16 h[2]; unsigned int u; } p;
    p.h[0] = (__bf16)a; p.h[1] = (__bf16)b;
    return p.u;                            // lowers to v_cvt_pk_bf16_f32
}

// A-matrix fragment (16x32 bf16, row-major tile in LDS, ld elements/row).
// Lanes 0-15: row M=lane, K chunks {0..7} and {16..23}.
// Lanes 16-31: row M=lane-16, K chunks {8..15} and {24..31}.
__device__ __forceinline__ bf16x16 load_fragA(const __bf16* base, int ld, int lane) {
    const int row = lane & 15;
    const int kb  = (lane < 16) ? 0 : 8;
    FragU f;
    f.u[0] = *(const u32x4*)(base + row * ld + kb);
    f.u[1] = *(const u32x4*)(base + row * ld + kb + 16);
    return f.v;
}

// B-matrix fragment (32x16 bf16). Tile stored K-transposed in LDS: [N][K].
// Lanes 0-15: col N=lane, K chunks {0..7} and {16..23}. Lanes 16-31: col, K chunks {8..15} and {24..31}.
__device__ __forceinline__ bf16x16 load_fragB(const __bf16* base, int ld, int lane) {
    const int col = lane & 15;
    const int kb  = (lane < 16) ? 0 : 8;
    FragU f;
    f.u[0] = *(const u32x4*)(base + col * ld + kb);
    f.u[1] = *(const u32x4*)(base + col * ld + kb + 16);
    return f.v;
}

__device__ __forceinline__ int region9(int gh, int gw) {
    int rh = (gh < 49) ? 0 : ((gh < 53) ? 1 : 2);
    int rw = (gw < 49) ? 0 : ((gw < 53) ? 1 : 2);
    return rh * 3 + rw;
}

// TDM: async 2D tile load (bf16), 128 rows x 32 elems, row stride 384 elems,
// into LDS rows padded to 40 elems (pad 4 DWORDs every 16 DWORDs).
// D# per CDNA5 ISA ch.8: group0 {count, lds_addr, global_addr, type=2},
// group1 {data_size=2B, pad_enable, pad_interval=16dw, pad_amount=4dw,
//         tensor_dim0=384, tensor_dim1=rows, tile_dim0=32, tile_dim1=128,
//         tensor_dim0_stride=384}, higher-dim groups unused (2D tile).
__device__ __forceinline__ void tdm_load_tile_bf16(const __bf16* gptr,
                                                   unsigned lds_off, int rows) {
    unsigned long long ga = (unsigned long long)gptr;
    u32x4 g0;
    g0[0] = 1u;                                  // count=1, user mode
    g0[1] = lds_off;                             // LDS byte address
    g0[2] = (unsigned)ga;                        // global_addr[31:0]
    g0[3] = (unsigned)(ga >> 32) | (2u << 30);   // global_addr[56:32] | type=2
    i32x8 g1;
    g1[0] = (1 << 16) | (1 << 20) | (3 << 22) | (3 << 25);
    g1[1] = (384 & 0xFFFF) << 16;                           // tensor_dim0 lo
    g1[2] = (int)((384u >> 16) | ((unsigned)(rows & 0xFFFF) << 16));
    g1[3] = (rows >> 16) | (32 << 16);                      // dim1 hi | tile_dim0
    g1[4] = 128;                                            // tile_dim1
    g1[5] = 384;                                            // tensor_dim0_stride lo
    g1[6] = 0; g1[7] = 0;
    i32x4 z4 = {};
    i32x8 z8 = {};
    __builtin_amdgcn_tensor_load_to_lds(g0, g1, z4, z4, z8, 0);
}

// ---------------------------------------------------------------------------
// Kernel 1: qkv = roll(x,-3,-3) @ qkv_w + qkv_b  -> bf16, windowed layout
// Output layout: [b][win][t][head][m][d];  index = dstRow[m] + t*18816 +
// head*1568 + d.  Block 128x128, 8 waves of 64x32 (4x2 tiles), K-step 32.
// ---------------------------------------------------------------------------
__global__ __launch_bounds__(256)
void qkv_kernel(const float* __restrict__ x, const float* __restrict__ qkv_w,
                const float* __restrict__ qkv_b, __bf16* __restrict__ qkv_out)
{
    __shared__ __align__(16) __bf16 As[128][40];
    __shared__ __align__(16) __bf16 Bt[128][40];
    __shared__ int rowSrc[128];
    __shared__ int dstRow[128];

    const int tid = threadIdx.x;
    const int m0  = blockIdx.x * 128;
    const int n0  = blockIdx.y * 128;

    if (tid < 128) {                       // per-block row decode tables
        int r  = m0 + tid;
        int b  = r / 3136, l = r % 3136;
        int hs = l / 56,   ws = l % 56;
        rowSrc[tid] = b * 3136 + ((hs + 3) % 56) * 56 + ((ws + 3) % 56);
        int win = (hs / 7) * 8 + (ws / 7);
        int mm  = (hs % 7) * 7 + (ws % 7);
        dstRow[tid] = b * 3612672 + win * 56448 + mm * 32;
    }

    const int wave = tid >> 5, lane = tid & 31;
    const int wm = (wave & 1) * 64;
    const int wn = (wave >> 1) * 32;

    f32x8 acc[4][2];
    #pragma unroll
    for (int i = 0; i < 4; ++i)
        #pragma unroll
        for (int j = 0; j < 2; ++j) { f32x8 z = {}; acc[i][j] = z; }

    for (int k0 = 0; k0 < 384; k0 += 32) {
        __syncthreads();
        #pragma unroll
        for (int i = 0; i < 8; ++i) {      // A: 128x32, f32x2 load -> packed b32
            int e = tid + i * 256;         // pair index
            int r = e >> 4, kk = (e & 15) * 2;
            f32x2 xv = *(const f32x2*)&x[(size_t)rowSrc[r] * 384 + k0 + kk];
            *(unsigned int*)&As[r][kk] = pack_bf16(xv.x, xv.y);
        }
        #pragma unroll
        for (int i = 0; i < 8; ++i) {      // B: 32x128, K-transposed, packed b32
            int e = tid + i * 256;
            int n = e & 127, kp = (e >> 7) * 2;
            float a0 = qkv_w[(size_t)(k0 + kp)     * 1152 + n0 + n];
            float a1 = qkv_w[(size_t)(k0 + kp + 1) * 1152 + n0 + n];
            *(unsigned int*)&Bt[n][kp] = pack_bf16(a0, a1);
        }
        __syncthreads();

        bf16x16 af[4], bf[2];
        #pragma unroll
        for (int mi = 0; mi < 4; ++mi) af[mi] = load_fragA(&As[wm + mi * 16][0], 40, lane);
        #pragma unroll
        for (int ni = 0; ni < 2; ++ni) bf[ni] = load_fragB(&Bt[wn + ni * 16][0], 40, lane);
        #pragma unroll
        for (int mi = 0; mi < 4; ++mi)
            #pragma unroll
            for (int ni = 0; ni < 2; ++ni)
                acc[mi][ni] = __builtin_amdgcn_wmma_f32_16x16x32_bf16(
                    false, af[mi], false, bf[ni], (short)0, acc[mi][ni], false, false);
    }

    const int mofs = (lane >> 4) * 8;
    #pragma unroll
    for (int ni = 0; ni < 2; ++ni) {
        int n    = n0 + wn + ni * 16 + (lane & 15);
        int t    = n / 384, rem = n % 384;
        int head = rem >> 5, d = rem & 31;
        int colOff = t * 18816 + head * 1568 + d;
        float bias = qkv_b[n];
        #pragma unroll
        for (int mi = 0; mi < 4; ++mi) {
            #pragma unroll
            for (int r = 0; r < 8; ++r) {
                int ml = wm + mi * 16 + mofs + r;
                qkv_out[(size_t)dstRow[ml] + colOff] = (__bf16)(acc[mi][ni][r] + bias);
            }
        }
    }
}

// ---------------------------------------------------------------------------
// Kernel 2: per (b, win, head) attention. M=49 padded to 64.
//   scores = rms(q)*s @ rms(k)^T + shift_mask + rel_bias ; softmax ; @ v
// Output: bf16 [shifted_row][head*32+d] rows for the projection GEMM.
// ---------------------------------------------------------------------------
__global__ __launch_bounds__(256)
void attn_kernel(const __bf16* __restrict__ qkv, const float* __restrict__ bias_table,
                 const float* __restrict__ qnw, const float* __restrict__ knw,
                 __bf16* __restrict__ out)
{
    __shared__ __align__(16) __bf16 Qs[64][40];
    __shared__ __align__(16) __bf16 Ks[64][40];
    __shared__ __align__(16) __bf16 Vt[32][72];
    __shared__ __align__(16) float  Sc[64][64];
    __shared__ __align__(16) __bf16 Ps[64][72];
    __shared__ __align__(16) float  BiasM[49][52];

    const int bid  = blockIdx.x;
    const int head = bid % 12;
    const int win  = (bid / 12) % 64;
    const int b    = bid / (12 * 64);
    const int wh = win >> 3, ww = win & 7;

    const size_t qbase = ((((size_t)b * 64 + win) * 3 + 0) * 12 + head) * (49 * 32);
    const size_t kbase = ((((size_t)b * 64 + win) * 3 + 1) * 12 + head) * (49 * 32);
    const size_t vbase = ((((size_t)b * 64 + win) * 3 + 2) * 12 + head) * (49 * 32);

    const int tid = threadIdx.x;
    const float scale = 0.17677669529663687f;   // hd^-0.5
    const float eps   = 1.1920929e-7f;          // finfo(f32).eps

    // --- shift-mask + rel-pos bias table (one gather per element)
    for (int e = tid; e < 49 * 49; e += 256) {
        int m = e / 49, n = e % 49;
        int mh = m / 7, mw = m % 7, nh = n / 7, nw = n % 7;
        int rm = region9(wh * 7 + mh, ww * 7 + mw);
        int rn = region9(wh * 7 + nh, ww * 7 + nw);
        float msk = (rm != rn) ? -100.f : 0.f;
        BiasM[m][n] = msk + bias_table[((mh - nh + 6) * 13 + (mw - nw + 6)) * 12 + head];
    }

    // --- load q,k (vector b128) with RMS norm; scale folded into q
    if (tid < 64) {
        int m = tid;
        if (m < 49) {
            Bf8U qc[4], kc[4];
            const u32x4* qp = (const u32x4*)(qkv + qbase + (size_t)m * 32);
            const u32x4* kp = (const u32x4*)(qkv + kbase + (size_t)m * 32);
            #pragma unroll
            for (int c = 0; c < 4; ++c) { qc[c].v = qp[c]; kc[c].v = kp[c]; }
            float sq = 0.f, sk = 0.f;
            #pragma unroll
            for (int c = 0; c < 4; ++c)
                #pragma unroll
                for (int j = 0; j < 8; ++j) {
                    float a = (float)qc[c].h[j], k2 = (float)kc[c].h[j];
                    sq += a * a; sk += k2 * k2;
                }
            float rq = rsqrtf(sq * (1.f / 32.f) + eps) * scale;
            float rk = rsqrtf(sk * (1.f / 32.f) + eps);
            #pragma unroll
            for (int c = 0; c < 4; ++c)
                #pragma unroll
                for (int j = 0; j < 8; j += 2) {
                    int d = c * 8 + j;
                    *(unsigned int*)&Qs[m][d] = pack_bf16(
                        (float)qc[c].h[j] * rq * qnw[d], (float)qc[c].h[j+1] * rq * qnw[d+1]);
                    *(unsigned int*)&Ks[m][d] = pack_bf16(
                        (float)kc[c].h[j] * rk * knw[d], (float)kc[c].h[j+1] * rk * knw[d+1]);
                }
        } else {
            #pragma unroll
            for (int d = 0; d < 32; d += 2) {
                *(unsigned int*)&Qs[m][d] = 0u;
                *(unsigned int*)&Ks[m][d] = 0u;
            }
        }
    }
    // --- load v transposed: Vt[d][m]  (one b128 chunk per thread)
    {
        int m = tid >> 2, db = (tid & 3) * 8;
        if (m < 49) {
            Bf8U vc;
            vc.v = *(const u32x4*)(qkv + vbase + (size_t)m * 32 + db);
            #pragma unroll
            for (int j = 0; j < 8; ++j) Vt[db + j][m] = vc.h[j];
        } else {
            #pragma unroll
            for (int j = 0; j < 8; ++j) Vt[db + j][m] = (__bf16)0.f;
        }
    }
    __syncthreads();

    const int wave = tid >> 5, lane = tid & 31;
    const int mofs = (lane >> 4) * 8;

    // --- scores = Q @ K^T : 4x4 tiles, 2 per wave (K rows give B^T layout)
    #pragma unroll
    for (int tix = 0; tix < 2; ++tix) {
        int t  = wave * 2 + tix;
        int mi = t >> 2, ni = t & 3;
        bf16x16 a  = load_fragA(&Qs[mi * 16][0], 40, lane);
        bf16x16 bb = load_fragB(&Ks[ni * 16][0], 40, lane);
        f32x8 c = {};
        c = __builtin_amdgcn_wmma_f32_16x16x32_bf16(false, a, false, bb, (short)0, c, false, false);
        int n = ni * 16 + (lane & 15);
        #pragma unroll
        for (int r = 0; r < 8; ++r) Sc[mi * 16 + mofs + r][n] = c[r];
    }
    __syncthreads();

    // --- fused bias + softmax (one thread per row; biases from LDS)
    if (tid < 64) {
        int m = tid;
        if (m < 49) {
            float mx = -1e30f;
            for (int n = 0; n < 49; ++n)
                mx = fmaxf(mx, Sc[m][n] + BiasM[m][n]);
            float sum = 0.f;
            for (int n = 0; n < 49; ++n) {
                float e = __expf(Sc[m][n] + BiasM[m][n] - mx);
                Sc[m][n] = e; sum += e;
            }
            float inv = 1.f / sum;
            for (int n = 0; n < 48; n += 2)
                *(unsigned int*)&Ps[m][n] = pack_bf16(Sc[m][n] * inv, Sc[m][n+1] * inv);
            Ps[m][48] = (__bf16)(Sc[m][48] * inv); Ps[m][49] = (__bf16)0.f;
            for (int n = 50; n < 64; n += 2) *(unsigned int*)&Ps[m][n] = 0u;
        } else {
            for (int n = 0; n < 64; n += 2) *(unsigned int*)&Ps[m][n] = 0u;
        }
    }
    __syncthreads();

    // --- out = P(64x64) @ V(64x32): 4x2 tiles, one per wave, K=64 in 2 steps
    {
        int mi = wave >> 1;
        int ni = wave & 1;
        f32x8 c = {};
        #pragma unroll
        for (int k0 = 0; k0 < 64; k0 += 32) {
            bf16x16 a  = load_fragA(&Ps[mi * 16][k0], 72, lane);
            bf16x16 bb = load_fragB(&Vt[ni * 16][k0], 72, lane);
            c = __builtin_amdgcn_wmma_f32_16x16x32_bf16(false, a, false, bb, (short)0, c, false, false);
        }
        int d = ni * 16 + (lane & 15);
        #pragma unroll
        for (int r = 0; r < 8; ++r) {
            int m = mi * 16 + mofs + r;
            if (m < 49) {
                int hs = wh * 7 + m / 7;
                int ws = ww * 7 + m % 7;
                size_t row = (size_t)b * 3136 + hs * 56 + ws;   // still shifted space
                out[row * 384 + head * 32 + d] = (__bf16)c[r];
            }
        }
    }
}

// ---------------------------------------------------------------------------
// Kernel 3: out = attn @ proj_w + proj_b, with roll(+3,+3) folded into store.
// A-tile fill uses the Tensor Data Mover (async, overlapped with B fill).
// ---------------------------------------------------------------------------
__global__ __launch_bounds__(256)
void proj_kernel(const __bf16* __restrict__ a_in, const float* __restrict__ proj_w,
                 const float* __restrict__ proj_b, float* __restrict__ out)
{
    __shared__ __align__(16) __bf16 As[128][40];
    __shared__ __align__(16) __bf16 Bt[128][40];
    __shared__ int dstRow[128];

    const int tid = threadIdx.x;
    const int m0  = blockIdx.x * 128;
    const int n0  = blockIdx.y * 128;

    if (tid < 128) {                       // unshift decode table (roll +3,+3)
        int m  = m0 + tid;
        int bb = m / 3136, l = m % 3136;
        int hs = l / 56,   ws = l % 56;
        dstRow[tid] = (bb * 3136 + ((hs + 3) % 56) * 56 + ((ws + 3) % 56)) * 384;
    }

    const int wave = tid >> 5, lane = tid & 31;
    const int wm = (wave & 1) * 64;
    const int wn = (wave >> 1) * 32;
    const unsigned ldsA = (unsigned)(unsigned long long)&As[0][0];

    f32x8 acc[4][2];
    #pragma unroll
    for (int i = 0; i < 4; ++i)
        #pragma unroll
        for (int j = 0; j < 2; ++j) { f32x8 z = {}; acc[i][j] = z; }

    for (int k0 = 0; k0 < 384; k0 += 32) {
        __syncthreads();
        if (tid == 0)                      // A: one TDM 2D-tile load (async)
            tdm_load_tile_bf16(a_in + (size_t)m0 * 384 + k0, ldsA, 100352);
        #pragma unroll
        for (int i = 0; i < 8; ++i) {      // B: 32x128 K-transposed, packed b32
            int e = tid + i * 256;
            int n = e & 127, kp = (e >> 7) * 2;
            float a0 = proj_w[(size_t)(k0 + kp)     * 384 + n0 + n];
            float a1 = proj_w[(size_t)(k0 + kp + 1) * 384 + n0 + n];
            *(unsigned int*)&Bt[n][kp] = pack_bf16(a0, a1);
        }
        if (tid == 0)
            __builtin_amdgcn_s_wait_tensorcnt(0);
        __syncthreads();

        bf16x16 af[4], bf[2];
        #pragma unroll
        for (int mi = 0; mi < 4; ++mi) af[mi] = load_fragA(&As[wm + mi * 16][0], 40, lane);
        #pragma unroll
        for (int ni = 0; ni < 2; ++ni) bf[ni] = load_fragB(&Bt[wn + ni * 16][0], 40, lane);
        #pragma unroll
        for (int mi = 0; mi < 4; ++mi)
            #pragma unroll
            for (int ni = 0; ni < 2; ++ni)
                acc[mi][ni] = __builtin_amdgcn_wmma_f32_16x16x32_bf16(
                    false, af[mi], false, bf[ni], (short)0, acc[mi][ni], false, false);
    }

    const int mofs = (lane >> 4) * 8;
    #pragma unroll
    for (int ni = 0; ni < 2; ++ni) {
        int n = n0 + wn + ni * 16 + (lane & 15);
        float bias = proj_b[n];
        #pragma unroll
        for (int mi = 0; mi < 4; ++mi) {
            #pragma unroll
            for (int r = 0; r < 8; ++r) {
                int ml = wm + mi * 16 + mofs + r;
                out[(size_t)dstRow[ml] + n] = acc[mi][ni][r] + bias;
            }
        }
    }
}

// ---------------------------------------------------------------------------
extern "C" void kernel_launch(void* const* d_in, const int* in_sizes, int n_in,
                              void* d_out, int out_size, void* d_ws, size_t ws_size,
                              hipStream_t stream)
{
    const float* x          = (const float*)d_in[0];
    const float* qkv_w      = (const float*)d_in[1];
    const float* qkv_b      = (const float*)d_in[2];
    const float* proj_w     = (const float*)d_in[3];
    const float* proj_b     = (const float*)d_in[4];
    const float* bias_table = (const float*)d_in[5];
    const float* q_norm_w   = (const float*)d_in[6];
    const float* k_norm_w   = (const float*)d_in[7];

    __bf16* qkv_ws = (__bf16*)d_ws;                               // 231 MB bf16
    const size_t qkv_elems = (size_t)32 * 64 * 3 * 12 * 49 * 32;  // [b][win][t][h][m][d]
    __bf16* attn_ws = qkv_ws + qkv_elems;                         // 77 MB bf16

    qkv_kernel <<<dim3(784, 9), 256, 0, stream>>>(x, qkv_w, qkv_b, qkv_ws);
    attn_kernel<<<dim3(32 * 64 * 12), 256, 0, stream>>>(qkv_ws, bias_table,
                                                        q_norm_w, k_norm_w, attn_ws);
    proj_kernel<<<dim3(784, 3), 256, 0, stream>>>(attn_ws, proj_w, proj_b, (float*)d_out);
}
